// S4D_90623809945982
// MI455X (gfx1250) — matrix operand order repaired and checked by reading, not compiled, not run on hardware
//
#include <hip/hip_runtime.h>

typedef __attribute__((ext_vector_type(16))) _Float16 v16h;
typedef __attribute__((ext_vector_type(8)))  float    v8f;

#define HDIM 1024
#define LDIM 2048
#define BDIM 4
#define NST  32      // N2 complex states
#define TCH  64      // chunk length
#define NCH  32      // chunks per batch (LDIM / TCH)
#define COLS 128     // BDIM * NCH matmul columns

// ---- LDS: ~57 KB with phase-overlaid unions (16B aligned for b128 LDS ops) ----
struct alignas(16) SMem {
  union R1 {                         // setup: Vandermonde powers A^t ; phase1 out: S^T
    struct { float apRe[65 * NST]; float apIm[65 * NST]; } a;   // 16640 B
    _Float16 ST[COLS * TCH];                                    // 16384 B  [col][comp]
  } r1;
  union R2 {                         // phase1 in: Toep, W_B ; scan out: X_start
    struct { _Float16 Toep[TCH * TCH]; _Float16 WB[TCH * TCH]; } m; // 16384 B
    _Float16 XT[COLS * TCH];                                        // [col][comp]
  } r2;
  _Float16 WC[TCH * TCH];            // 8192 B  [t][comp]
  _Float16 UT[COLS * TCH];           // 16384 B [col][s]  (transposed u chunks)
  float cmRe[NST], cmIm[NST];        // Cmod
  float a64Re[NST], a64Im[NST];      // A^T for the scan
  float dtAre[NST], dtAim[NST];
  float Kv[TCH];                     // conv kernel taps K[0..63]
};

// A-operand fragment (16x32 f16), row-major LDS [M][K], per ISA 7.12.2:
// lanes 0-15: M=lane, VGPR0-3 K=0..7, VGPR4-7 K=16..23 ; lanes 16-31: +8 on K groups
// Each half is 16 contiguous, 16B-aligned bytes -> two ds_load_b128.
__device__ inline v16h load_frag_a(const _Float16* lds, int row0, int k0, int ldk, int lane) {
  union { v16h v; uint4 q[2]; } f;
  const int m = row0 + (lane & 15);
  const int g = (lane & 16) ? 8 : 0;
  const _Float16* p = lds + m * ldk + k0 + g;
  f.q[0] = *(const uint4*)(p);
  f.q[1] = *(const uint4*)(p + 16);
  return f.v;
}

// B-operand fragment (32x16 f16) from transposed LDS [N][K]:
// lanes 0-15 hold K=0..7,16..23, lanes 16-31 hold K=8..15,24..31 (matches A) -> two ds_load_b128.
__device__ inline v16h load_frag_b(const _Float16* ldsT, int col0, int k0, int ldk, int lane) {
  union { v16h v; uint4 q[2]; } f;
  const int n = col0 + (lane & 15);
  const int g = (lane & 16) ? 8 : 0;
  const _Float16* p = ldsT + n * ldk + k0 + g;
  f.q[0] = *(const uint4*)(p);
  f.q[1] = *(const uint4*)(p + 16);
  return f.v;
}

__global__ __launch_bounds__(256) void s4d_chunkscan_kernel(
    const float* __restrict__ u,  const float* __restrict__ w_re, const float* __restrict__ w_im,
    const float* __restrict__ Cr, const float* __restrict__ Ci,
    const float* __restrict__ Dp, const float* __restrict__ dtp,
    float* __restrict__ out)
{
  __shared__ SMem sm;
  const int h    = blockIdx.x;
  const int tid  = threadIdx.x;
  const int lane = tid & 31;
  const int wave = tid >> 5;

  // ---- prefetch this channel's u column toward L2 while setup math runs ----
  for (int i = tid; i < COLS * TCH; i += 256) {
    const int col = i >> 6, s = i & 63;
    const int b = col >> 5, c = col & 31;
    const size_t gidx = ((size_t)b * LDIM + (size_t)(c * TCH + s)) * HDIM + (size_t)h;
    __builtin_prefetch(&u[gidx], 0, 1);
  }

  const float dtv = dtp[h];
  const float Dh  = Dp[h];

  // ---- setup A: per-state dtA, A=exp(dtA), Cmod = C*(A-1)/w ----
  if (tid < NST) {
    const int n = tid;
    const float wr = w_re[n], wi = w_im[n];
    const float ar = wr * dtv, ai = wi * dtv;
    sm.dtAre[n] = ar; sm.dtAim[n] = ai;
    float sn, cs;
    const float er = __expf(ar);
    __sincosf(ai, &sn, &cs);
    const float Er = er * cs, Ei = er * sn;
    const float nr = Er - 1.0f, ni = Ei;
    const float inv = 1.0f / (wr * wr + wi * wi);
    const float qr = (nr * wr + ni * wi) * inv;      // (A-1)/w
    const float qi = (ni * wr - nr * wi) * inv;
    const float cr = Cr[h * NST + n], ci = Ci[h * NST + n];
    sm.cmRe[n] = cr * qr - ci * qi;
    sm.cmIm[n] = cr * qi + ci * qr;
  }
  __syncthreads();

  // ---- setup B: Vandermonde powers A^t, t=0..64 (direct exp for accuracy) ----
  for (int i = tid; i < 65 * NST; i += 256) {
    const int t = i >> 5, n = i & 31;
    const float pr = (float)t * sm.dtAre[n];
    const float pi = (float)t * sm.dtAim[n];
    float sn, cs;
    const float er = __expf(pr);
    __sincosf(pi, &sn, &cs);
    const float Re = er * cs, Im = er * sn;
    sm.r1.a.apRe[i] = Re; sm.r1.a.apIm[i] = Im;
    if (t == 64) { sm.a64Re[n] = Re; sm.a64Im[n] = Im; }
  }
  __syncthreads();

  // ---- setup C: conv taps K[t] = 2 Re sum_n Cmod_n A_n^t ----
  if (tid < TCH) {
    float acc = 0.f;
#pragma unroll 8
    for (int n = 0; n < NST; ++n)
      acc += sm.cmRe[n] * sm.r1.a.apRe[tid * NST + n] - sm.cmIm[n] * sm.r1.a.apIm[tid * NST + n];
    sm.Kv[tid] = 2.0f * acc;
  }
  __syncthreads();

  // ---- setup D: build f16 operand matrices Toep, W_B, W_C ----
  for (int i = tid; i < TCH * TCH; i += 256) {
    const int r = i >> 6, s = i & 63;
    // Toep[t][s] = K[t-s] for s<=t
    sm.r2.m.Toep[i] = (s <= r) ? (_Float16)sm.Kv[r - s] : (_Float16)0.f;
    // W_B[comp r][s]: Re/Im of A^{63-s} for state r%32
    const int nb = r & 31;
    const float wb = (r < NST) ? sm.r1.a.apRe[(63 - s) * NST + nb]
                               : sm.r1.a.apIm[(63 - s) * NST + nb];
    sm.r2.m.WB[i] = (_Float16)wb;
    // W_C[t r][comp s]: 2Re / -2Im of Cmod*A^{t+1} for state s%32
    const int nc = s & 31;
    const float gr = sm.cmRe[nc] * sm.r1.a.apRe[(r + 1) * NST + nc]
                   - sm.cmIm[nc] * sm.r1.a.apIm[(r + 1) * NST + nc];
    const float gi = sm.cmRe[nc] * sm.r1.a.apIm[(r + 1) * NST + nc]
                   + sm.cmIm[nc] * sm.r1.a.apRe[(r + 1) * NST + nc];
    sm.WC[i] = (_Float16)((s < NST) ? 2.0f * gr : -2.0f * gi);
  }

  // ---- load u for this channel: UT[col=(b,chunk)][s] (L2 amortizes stride-H gather) ----
  for (int i = tid; i < COLS * TCH; i += 256) {
    const int col = i >> 6, s = i & 63;
    const int b = col >> 5, c = col & 31;
    const size_t gidx = ((size_t)b * LDIM + (size_t)(c * TCH + s)) * HDIM + (size_t)h;
    sm.UT[i] = (_Float16)u[gidx];
  }
  __syncthreads();

  // ---- phase 1: Y_intra = Toep@U (kept in accumulators), S = W_B@U (to LDS, transposed) ----
  const int mt  = wave >> 1;           // output row tile 0..3
  const int ntb = (wave & 1) * 4;      // first of 4 column tiles
  v8f yi[4];

  const v16h aT0 = load_frag_a(sm.r2.m.Toep, mt * 16, 0,  TCH, lane);
  const v16h aT1 = load_frag_a(sm.r2.m.Toep, mt * 16, 32, TCH, lane);
  const v16h aB0 = load_frag_a(sm.r2.m.WB,   mt * 16, 0,  TCH, lane);
  const v16h aB1 = load_frag_a(sm.r2.m.WB,   mt * 16, 32, TCH, lane);

#pragma unroll
  for (int q = 0; q < 4; ++q) {
    const int nt = ntb + q;
    const v16h b0 = load_frag_b(sm.UT, nt * 16, 0,  TCH, lane);
    const v16h b1 = load_frag_b(sm.UT, nt * 16, 32, TCH, lane);
    v8f z = {};
    v8f acc = __builtin_amdgcn_wmma_f32_16x16x32_f16(false, aT0, false, b0, (short)0, z,   false, false);
    acc     = __builtin_amdgcn_wmma_f32_16x16x32_f16(false, aT1, false, b1, (short)0, acc, false, false);
    yi[q] = acc;
    v8f sa  = __builtin_amdgcn_wmma_f32_16x16x32_f16(false, aB0, false, b0, (short)0, z,   false, false);
    sa      = __builtin_amdgcn_wmma_f32_16x16x32_f16(false, aB1, false, b1, (short)0, sa,  false, false);
    // store S transposed: lane's 8 rows are contiguous halves -> one b128 store
    const int colb = nt * 16 + (lane & 15);
    const int rb   = mt * 16 + ((lane & 16) ? 8 : 0);
    union { uint4 q4; _Float16 hv[8]; } pk;
#pragma unroll
    for (int r = 0; r < 8; ++r) pk.hv[r] = (_Float16)sa[r];
    *(uint4*)(sm.r1.ST + colb * TCH + rb) = pk.q4;
  }
  __syncthreads();

  // ---- phase 2: sequential inter-chunk scan, 128 independent complex sequences ----
  if (tid < BDIM * NST) {
    const int b = tid >> 5, n = tid & 31;
    float xr = 0.f, xi = 0.f;
    const float ar = sm.a64Re[n], ai = sm.a64Im[n];
    for (int c = 0; c < NCH; ++c) {
      const int col = b * NCH + c;
      sm.r2.XT[col * TCH + n]       = (_Float16)xr;   // X_start for chunk c
      sm.r2.XT[col * TCH + NST + n] = (_Float16)xi;
      const float sr = (float)sm.r1.ST[col * TCH + n];
      const float si = (float)sm.r1.ST[col * TCH + NST + n];
      const float nxr = ar * xr - ai * xi + sr;       // x <- A^T * x + S
      const float nxi = ar * xi + ai * xr + si;
      xr = nxr; xi = nxi;
    }
  }
  __syncthreads();

  // ---- phase 3: Y += W_C @ X_start (accumulate into yi), add skip, store ----
  const v16h aC0 = load_frag_a(sm.WC, mt * 16, 0,  TCH, lane);
  const v16h aC1 = load_frag_a(sm.WC, mt * 16, 32, TCH, lane);
#pragma unroll
  for (int q = 0; q < 4; ++q) {
    const int nt = ntb + q;
    const v16h b0 = load_frag_b(sm.r2.XT, nt * 16, 0,  TCH, lane);
    const v16h b1 = load_frag_b(sm.r2.XT, nt * 16, 32, TCH, lane);
    v8f acc = __builtin_amdgcn_wmma_f32_16x16x32_f16(false, aC0, false, b0, (short)0, yi[q], false, false);
    acc     = __builtin_amdgcn_wmma_f32_16x16x32_f16(false, aC1, false, b1, (short)0, acc,   false, false);
    const int colb = nt * 16 + (lane & 15);
    const int rb   = mt * 16 + ((lane & 16) ? 8 : 0);
    const int b    = colb >> 5, c = colb & 31;
#pragma unroll
    for (int r = 0; r < 8; ++r) {
      const int t = rb + r;
      const int l = c * TCH + t;
      const float uval = (float)sm.UT[colb * TCH + t];
      out[((size_t)b * LDIM + (size_t)l) * HDIM + (size_t)h] = acc[r] + Dh * uval;
    }
  }
}

extern "C" void kernel_launch(void* const* d_in, const int* in_sizes, int n_in,
                              void* d_out, int out_size, void* d_ws, size_t ws_size,
                              hipStream_t stream) {
  const float* u    = (const float*)d_in[0];
  const float* w_re = (const float*)d_in[1];
  const float* w_im = (const float*)d_in[2];
  const float* C_re = (const float*)d_in[3];
  const float* C_im = (const float*)d_in[4];
  const float* D    = (const float*)d_in[5];
  const float* dt   = (const float*)d_in[6];
  float* out = (float*)d_out;
  s4d_chunkscan_kernel<<<dim3(HDIM), dim3(256), 0, stream>>>(u, w_re, w_im, C_re, C_im, D, dt, out);
}
